// TogepiMultiHeadAttention_78743930405234
// MI455X (gfx1250) — matrix operand order repaired and checked by reading, not compiled, not run on hardware
//
#include <hip/hip_runtime.h>
#include <math.h>

#define PI_F 3.14159265358979323846f

namespace {

constexpr int kB = 16;
constexpr int kL = 1024;
constexpr int kD = 768;
constexpr int kH = 12;
constexpr int kPHD = 64;
constexpr int kTaps = 2047;  // 2L-1
constexpr int kF = 2048;     // pow2 FFT length (>= 2L-1, alias free for causal part)

// GEMM tiling: 256(M) x 128(N) workgroup tile, 8 waves as 4(M) x 2(N),
// each wave owns a 64x64 tile = 4x4 WMMA accumulators (1:1 ds_load:wmma).
constexpr int kTM = 256;
constexpr int kTN = 128;

typedef __attribute__((ext_vector_type(16))) __bf16 v16bf;
typedef __attribute__((ext_vector_type(8)))  __bf16 v8bf;
typedef __attribute__((ext_vector_type(8)))  float  v8f;
typedef int v4i_gcc __attribute__((vector_size(16)));  // matches builtin pointee type

struct cplx { float re, im; };
__device__ __forceinline__ cplx cmul(cplx a, cplx b) {
  return cplx{a.re * b.re - a.im * b.im, a.re * b.im + a.im * b.re};
}

// ---------------------------------------------------------------------------
// CDNA5 async global->LDS path (guarded; falls back to VGPR copies)
// ---------------------------------------------------------------------------
#if defined(__has_builtin)
#if __has_builtin(__builtin_amdgcn_global_load_async_to_lds_b128) && \
    __has_builtin(__builtin_amdgcn_s_wait_asynccnt)
#define ATHENA_ASYNC_LDS 1
#endif
#endif
#ifndef ATHENA_ASYNC_LDS
#define ATHENA_ASYNC_LDS 0
#endif

// copy one 16-byte chunk global -> LDS (async when available)
__device__ __forceinline__ void copy_chunk16(const __bf16* __restrict__ g, __bf16* l) {
#if ATHENA_ASYNC_LDS
  __builtin_amdgcn_global_load_async_to_lds_b128((v4i_gcc*)g, (v4i_gcc*)l, 0, 0);
#else
  *(v8bf*)l = *(const v8bf*)g;
#endif
}

__device__ __forceinline__ void wait_async() {
#if ATHENA_ASYNC_LDS
  __builtin_amdgcn_s_wait_asynccnt(0);
#endif
}

// ---------------------------------------------------------------------------
// WMMA fragment loads from LDS (layouts per CDNA5 ISA 7.12.2, wave32)
// A tile in LDS: [row 0..15][k 0..31] bf16, 64B rows.
//   lanes 0-15: M=lane, K={0..7,16..23}; lanes 16-31: M=lane-16, K={8..15,24..31}
// B tile in LDS (k-panel packed): [n 0..15][k 0..31] bf16, 64B rows.
//   lanes 0-15: N=lane, K={0..7,16..23}; lanes 16-31: N=lane-16, K={8..15,24..31}
// ---------------------------------------------------------------------------
__device__ __forceinline__ v16bf load_a_frag(const __bf16* base, int lane) {
  const int row = lane & 15;
  const int kb = (lane >> 4) << 3;  // 0 or 8
  const __bf16* q = base + row * 32 + kb;
  union { v16bf v; v8bf h[2]; } u;
  u.h[0] = *(const v8bf*)(q);
  u.h[1] = *(const v8bf*)(q + 16);
  return u.v;
}

__device__ __forceinline__ v16bf load_b_frag(const __bf16* base, int lane) {
  const int n = lane & 15;
  const int kb = (lane >> 4) << 3;  // 0 or 8
  const __bf16* q = base + n * 32 + kb;
  union { v16bf v; v8bf h[2]; } u;
  u.h[0] = *(const v8bf*)(q);
  u.h[1] = *(const v8bf*)(q + 16);
  return u.v;
}

// ---------------------------------------------------------------------------
// Tiled bf16 WMMA GEMM: C[m,n] (+=) sum_k A[m,k] * Bp[k,n] (+ bias[n])
// A: row-major bf16 (M x K). Bp: k-panel packed bf16: element (k,n) at
//    Bp[(k/32)*N*32 + n*32 + (k%32)]. C: f32 row-major (M x N).
// Workgroup: 256 threads = 8 waves; tile kTM x kTN; wave tile 64x64.
// M multiple of 256, N multiple of 128, K multiple of 32 (true for all uses).
// ---------------------------------------------------------------------------
__global__ __launch_bounds__(256) void gemm_bf16_wmma(
    const __bf16* __restrict__ A, long long strideA,
    const __bf16* __restrict__ Bp, long long strideB,
    float* __restrict__ C, long long strideC,
    const float* __restrict__ bias,
    int M, int N, int K, int accumulate) {
  __shared__ __align__(64) __bf16 lA[2][kTM][32];
  __shared__ __align__(64) __bf16 lB[2][kTN][32];

  const int n0 = blockIdx.x * kTN;
  const int m0 = blockIdx.y * kTM;
  const __bf16* Ab = A + (size_t)blockIdx.z * strideA;
  const __bf16* Bb = Bp + (size_t)blockIdx.z * strideB;
  float* Cb = C + (size_t)blockIdx.z * strideC;

  const int tid = threadIdx.x;
  const int lane = tid & 31;
  const int wid = tid >> 5;
  const int wm = wid & 3;   // 4 waves across M (64 rows each)
  const int wn = wid >> 2;  // 2 waves across N (64 cols each)

  v8f acc[4][4] = {};

  const int nk = K >> 5;
  constexpr int kChunks = (kTM + kTN) * 4;  // 16B chunks per k-tile

  auto copy_tile = [&](int kt, int buf) {
#pragma unroll
    for (int it = 0; it < kChunks / 256; ++it) {
      const int c = tid + it * 256;
      if (c < kTM * 4) {  // uniform per iteration: it<4 -> A rows, else B rows
        const int row = c >> 2;
        const int part = (c & 3) << 3;
        copy_chunk16(Ab + (size_t)(m0 + row) * K + (kt << 5) + part,
                     &lA[buf][row][part]);
      } else {
        const int c2 = c - kTM * 4;
        const int row = c2 >> 2;
        const int part = (c2 & 3) << 3;
        copy_chunk16(Bb + ((size_t)kt * N + (n0 + row)) * 32 + part,
                     &lB[buf][row][part]);
      }
    }
  };

  copy_tile(0, 0);
  wait_async();
  __syncthreads();

  for (int kt = 0; kt < nk; ++kt) {
    const int buf = kt & 1;
    if (kt + 1 < nk) {
      copy_tile(kt + 1, buf ^ 1);
      if (kt + 2 < nk) {
        __builtin_prefetch(Ab + (size_t)(m0 + tid) * K + ((kt + 2) << 5), 0, 0);
      }
    }

    const __bf16* aBase = &lA[buf][wm * 64][0];
    const __bf16* bBase = &lB[buf][wn * 64][0];
    v16bf af[4];
    v16bf bf2[4];
#pragma unroll
    for (int mi = 0; mi < 4; ++mi) af[mi] = load_a_frag(aBase + mi * 16 * 32, lane);
#pragma unroll
    for (int nj = 0; nj < 4; ++nj) bf2[nj] = load_b_frag(bBase + nj * 16 * 32, lane);

#pragma unroll
    for (int mi = 0; mi < 4; ++mi)
#pragma unroll
      for (int nj = 0; nj < 4; ++nj)
        acc[mi][nj] = __builtin_amdgcn_wmma_f32_16x16x32_bf16(
            false, af[mi], false, bf2[nj], (short)0, acc[mi][nj], false, false);

    wait_async();
    __syncthreads();
  }

  // epilogue: C/D layout: VGPR r -> M = r (lanes 0-15) or 8+r (lanes 16-31), N = lane%16
#pragma unroll
  for (int mi = 0; mi < 4; ++mi) {
#pragma unroll
    for (int nj = 0; nj < 4; ++nj) {
      const int n = n0 + wn * 64 + nj * 16 + (lane & 15);
      const int mb = m0 + wm * 64 + mi * 16 + ((lane >> 4) << 3);
      const float bv = bias ? bias[n] : 0.0f;
#pragma unroll
      for (int r = 0; r < 8; ++r) {
        const size_t idx = (size_t)(mb + r) * N + n;
        float v = acc[mi][nj][r] + bv;
        if (accumulate) v += Cb[idx];
        Cb[idx] = v;
      }
    }
  }
}

// ---------------------------------------------------------------------------
// Conversions / packing
// ---------------------------------------------------------------------------
__global__ __launch_bounds__(256) void convert_bf16(const float* __restrict__ in,
                                                    __bf16* __restrict__ out) {
  const size_t i = (size_t)blockIdx.x * 256 + threadIdx.x;
  out[i] = (__bf16)in[i];
}

// W (N x K row-major f32) -> Bp packed bf16 representing B[k][n] = W[n][k]
__global__ __launch_bounds__(256) void pack_wT(const float* __restrict__ W,
                                               __bf16* __restrict__ Bp) {
  const size_t gid = (size_t)blockIdx.x * 256 + threadIdx.x;  // over 768*768
  const int n = (int)(gid / kD);
  const int k = (int)(gid % kD);
  Bp[((size_t)(k >> 5) * kD + n) * 32 + (k & 31)] = (__bf16)W[gid];
}

// x (B,L,D f32) -> per-batch packed bf16 B[k=seq][n=d]
__global__ __launch_bounds__(256) void pack_x(const float* __restrict__ x,
                                              __bf16* __restrict__ Bp) {
  const size_t gid = (size_t)blockIdx.x * 256 + threadIdx.x;  // over B*L*D
  const int n = (int)(gid % kD);
  const int t = (int)((gid / kD) % kL);
  const int b = (int)(gid / ((size_t)kD * kL));
  Bp[(((size_t)b * 32 + (t >> 5)) * kD + n) * 32 + (t & 31)] = (__bf16)x[gid];
}

// ---------------------------------------------------------------------------
// Suffix sums along seq: S[b,s,d] = sum_{j>s} x[b,j,d]
// ---------------------------------------------------------------------------
__global__ __launch_bounds__(256) void suffix_sum(const float* __restrict__ x,
                                                  float* __restrict__ S) {
  const int d = blockIdx.y * 256 + threadIdx.x;
  const int b = blockIdx.x;
  const float* xb = x + (size_t)b * kL * kD;
  float* Sb = S + (size_t)b * kL * kD;
  float acc = 0.0f;
  for (int s = kL - 1; s >= 0; --s) {
    Sb[(size_t)s * kD + d] = acc;
    acc += xb[(size_t)s * kD + d];
  }
}

// ---------------------------------------------------------------------------
// PSF softmax over 2047 taps per (h,e); taps >= L are masked-to-zero BEFORE
// softmax so they each contribute exp(0)=1 to Z and the tail tap value is 1/Z.
// ---------------------------------------------------------------------------
__global__ __launch_bounds__(256) void psf_softmax(const float* __restrict__ psf,
                                                   float* __restrict__ psfL,
                                                   float* __restrict__ cvec) {
  __shared__ float red[256];
  const int he = blockIdx.x;  // h*64 + e
  const int h = he >> 6;
  const int e = he & 63;
  const int tid = threadIdx.x;
  float vals[4];
  float lsum = 0.0f;
#pragma unroll
  for (int i = 0; i < 4; ++i) {
    const int t = tid + i * 256;  // t < 1024
    const float ev = expf(psf[((size_t)h * kTaps + t) * kPHD + e]);
    vals[i] = ev;
    lsum += ev;
  }
  red[tid] = lsum;
  __syncthreads();
  for (int off = 128; off > 0; off >>= 1) {
    if (tid < off) red[tid] += red[tid + off];
    __syncthreads();
  }
  const float invZ = 1.0f / (red[0] + 1023.0f);  // 1023 masked taps, exp(0)=1 each
#pragma unroll
  for (int i = 0; i < 4; ++i) {
    const int t = tid + i * 256;
    psfL[((size_t)h * kL + t) * kPHD + e] = vals[i] * invZ;
  }
  if (tid == 0) cvec[he] = invZ;
}

// ---------------------------------------------------------------------------
// DFT-64 along head_dim (circular axis). One block (64 threads) per row.
// ---------------------------------------------------------------------------
__global__ __launch_bounds__(64) void dft64_x(const float* __restrict__ x,
                                              cplx* __restrict__ Xc) {
  __shared__ float xr[64];
  __shared__ cplx tw[64];
  const int g = blockIdx.x;  // (b*H + h)*L + s
  const int s = g % kL;
  const int bh = g / kL;
  const int h = bh % kH;
  const int b = bh / kH;
  const int tid = threadIdx.x;
  float sv, cv;
  __sincosf(-2.0f * PI_F * (float)tid / 64.0f, &sv, &cv);
  tw[tid] = cplx{cv, sv};
  xr[tid] = x[((size_t)(b * kL + s)) * kD + h * kPHD + tid];
  __syncthreads();
  float ar = 0.0f, ai = 0.0f;
#pragma unroll
  for (int e = 0; e < 64; ++e) {
    const cplx w = tw[(e * tid) & 63];
    const float v = xr[e];
    ar += v * w.re;
    ai += v * w.im;
  }
  Xc[((size_t)bh * 64 + tid) * kL + s] = cplx{ar, ai};
}

__global__ __launch_bounds__(64) void dft64_psf(const float* __restrict__ psfL,
                                                cplx* __restrict__ Pf1) {
  __shared__ float xr[64];
  __shared__ cplx tw[64];
  const int g = blockIdx.x;  // h*L + t
  const int t = g % kL;
  const int h = g / kL;
  const int tid = threadIdx.x;
  float sv, cv;
  __sincosf(-2.0f * PI_F * (float)tid / 64.0f, &sv, &cv);
  tw[tid] = cplx{cv, sv};
  xr[tid] = psfL[((size_t)h * kL + t) * kPHD + tid];
  __syncthreads();
  float ar = 0.0f, ai = 0.0f;
#pragma unroll
  for (int e = 0; e < 64; ++e) {
    const cplx w = tw[(e * tid) & 63];
    const float v = xr[e];
    ar += v * w.re;
    ai += v * w.im;
  }
  Pf1[((size_t)h * 64 + tid) * kL + t] = cplx{ar, ai};
}

// ---------------------------------------------------------------------------
// In-LDS radix-2 FFT, N=2048, 256 threads. inverse => +angle (scale by caller).
// ---------------------------------------------------------------------------
__device__ void fft2048(cplx* buf, int tid, bool inverse) {
  // bit reversal (11 bits)
  for (int i = tid; i < kF; i += 256) {
    const int j = (int)(__brev((unsigned)i) >> (32 - 11));
    if (j > i) {
      const cplx t = buf[i];
      buf[i] = buf[j];
      buf[j] = t;
    }
  }
  __syncthreads();
  const float sgn = inverse ? 2.0f : -2.0f;
  for (int st = 1; st <= 11; ++st) {
    const int m = 1 << st;
    const int half = m >> 1;
    for (int bfy = tid; bfy < (kF >> 1); bfy += 256) {
      const int grp = bfy / half;
      const int pos = bfy % half;
      const int i0 = grp * m + pos;
      const int i1 = i0 + half;
      float sw, cw;
      __sincosf(sgn * PI_F * (float)pos / (float)m, &sw, &cw);
      const cplx w{cw, sw};
      const cplx u = buf[i0];
      const cplx t = cmul(w, buf[i1]);
      buf[i0] = cplx{u.re + t.re, u.im + t.im};
      buf[i1] = cplx{u.re - t.re, u.im - t.im};
    }
    __syncthreads();
  }
}

__global__ __launch_bounds__(256) void fft_seq_psf(const cplx* __restrict__ Pf1,
                                                   cplx* __restrict__ Pfull) {
  __shared__ cplx buf[kF];
  const int col = blockIdx.x;  // h*64 + fe
  const int tid = threadIdx.x;
  const cplx* src = Pf1 + (size_t)col * kL;
  for (int i = tid; i < kF; i += 256) buf[i] = (i < kL) ? src[i] : cplx{0.0f, 0.0f};
  __syncthreads();
  fft2048(buf, tid, false);
  cplx* dst = Pfull + (size_t)col * kF;
  for (int i = tid; i < kF; i += 256) dst[i] = buf[i];
}

__global__ __launch_bounds__(256) void fft_seq_x(cplx* __restrict__ Xc,
                                                 const cplx* __restrict__ Pfull) {
  __shared__ cplx buf[kF];
  const int col = blockIdx.x;          // (b*H + h)*64 + fe
  const int hfe = col % (kH * 64);     // h*64 + fe
  const int tid = threadIdx.x;
  cplx* mine = Xc + (size_t)col * kL;
  for (int i = tid; i < kF; i += 256) buf[i] = (i < kL) ? mine[i] : cplx{0.0f, 0.0f};
  __syncthreads();
  fft2048(buf, tid, false);
  const cplx* P = Pfull + (size_t)hfe * kF;
  for (int i = tid; i < kF; i += 256) buf[i] = cmul(buf[i], P[i]);
  __syncthreads();
  fft2048(buf, tid, true);
  const float inv = 1.0f / (float)kF;
  for (int i = tid; i < kL; i += 256) mine[i] = cplx{buf[i].re * inv, buf[i].im * inv};
}

// ---------------------------------------------------------------------------
// Inverse DFT-64 along head_dim + tail (c circularly convolved with suffix
// sums) + exact GELU; emit bf16 activations row-major for the post GEMM.
// ---------------------------------------------------------------------------
__global__ __launch_bounds__(64) void conv_finish(const cplx* __restrict__ Yc,
                                                  const float* __restrict__ S,
                                                  const float* __restrict__ cvec,
                                                  __bf16* __restrict__ convAct) {
  __shared__ cplx yr[64];
  __shared__ float Sr[64];
  __shared__ float cr[64];
  __shared__ cplx twi[64];
  const int g = blockIdx.x;  // (b*H + h)*L + s
  const int s = g % kL;
  const int bh = g / kL;
  const int h = bh % kH;
  const int b = bh / kH;
  const int d = threadIdx.x;
  float sv, cv;
  __sincosf(2.0f * PI_F * (float)d / 64.0f, &sv, &cv);
  twi[d] = cplx{cv, sv};
  yr[d] = Yc[((size_t)bh * 64 + d) * kL + s];
  Sr[d] = S[((size_t)(b * kL + s)) * kD + h * kPHD + d];
  cr[d] = cvec[h * 64 + d];
  __syncthreads();
  float acc = 0.0f;
#pragma unroll
  for (int fe = 0; fe < 64; ++fe) {
    const cplx w = twi[(d * fe) & 63];
    acc += yr[fe].re * w.re - yr[fe].im * w.im;
  }
  acc *= (1.0f / 64.0f);
  float tail = 0.0f;
#pragma unroll
  for (int e = 0; e < 64; ++e) tail += cr[e] * Sr[(d - e) & 63];
  const float v = acc + tail;
  const float gelu = 0.5f * v * (1.0f + erff(v * 0.70710678118654752f));
  convAct[((size_t)(b * kL + s)) * kD + h * kPHD + d] = (__bf16)gelu;
}

// ---------------------------------------------------------------------------
// Spectral norm (1 power iteration, u0 = ones/sqrt(L)) + tril + bf16 scale
// ---------------------------------------------------------------------------
__device__ __forceinline__ float block_reduce_sum256(float v, float* red) {
  const int tid = threadIdx.x;
  red[tid] = v;
  __syncthreads();
  for (int off = 128; off > 0; off >>= 1) {
    if (tid < off) red[tid] += red[tid + off];
    __syncthreads();
  }
  const float r = red[0];
  __syncthreads();
  return r;
}

__global__ __launch_bounds__(256) void sn_colsum(const float* __restrict__ W,
                                                 float* __restrict__ v) {
  const int t = blockIdx.x * 256 + threadIdx.x;
  float s = 0.0f;
  for (int r = 0; r < kL; ++r) s += W[(size_t)r * kL + t];
  v[t] = s * 0.03125f;  // times 1/sqrt(1024)
}

__global__ __launch_bounds__(256) void sn_normalize_v(const float* __restrict__ v,
                                                      float* __restrict__ vhat) {
  __shared__ float red[256];
  const int tid = threadIdx.x;
  float ls = 0.0f;
  for (int i = tid; i < kL; i += 256) ls += v[i] * v[i];
  const float ss = block_reduce_sum256(ls, red);
  const float inv = 1.0f / (sqrtf(ss) + 1e-12f);
  for (int i = tid; i < kL; i += 256) vhat[i] = v[i] * inv;
}

__global__ __launch_bounds__(256) void sn_rowdot(const float* __restrict__ W,
                                                 const float* __restrict__ vhat,
                                                 float* __restrict__ w2) {
  __shared__ float red[256];
  const int srow = blockIdx.x;
  const int tid = threadIdx.x;
  float ls = 0.0f;
  for (int t = tid; t < kL; t += 256) ls += W[(size_t)srow * kL + t] * vhat[t];
  const float r = block_reduce_sum256(ls, red);
  if (tid == 0) w2[srow] = r;
}

__global__ __launch_bounds__(256) void sn_sigma(const float* __restrict__ w2,
                                                float* __restrict__ sigma) {
  __shared__ float red[256];
  const int tid = threadIdx.x;
  float ls = 0.0f;
  for (int i = tid; i < kL; i += 256) ls += w2[i] * w2[i];
  const float n2 = block_reduce_sum256(ls, red);
  if (tid == 0) sigma[0] = n2 / (sqrtf(n2) + 1e-12f);  // u_hat . (W v_hat)
}

__global__ __launch_bounds__(256) void sp_build(const float* __restrict__ W,
                                                const float* __restrict__ sigma,
                                                __bf16* __restrict__ sp) {
  const size_t i = (size_t)blockIdx.x * 256 + threadIdx.x;  // over L*L
  const int s = (int)(i >> 10);
  const int t = (int)(i & 1023);
  const float inv = 1.0f / sigma[0];
  sp[i] = (__bf16)((t <= s) ? W[i] * inv : 0.0f);
}

// ---------------------------------------------------------------------------
// Fused residual + LayerNorm (eps=1e-12): out = LN(accb + emb) * g + b
// One block (256 threads) per (b,s) row of 768.
// ---------------------------------------------------------------------------
__global__ __launch_bounds__(256) void ln_final(const float* __restrict__ accb,
                                                const float* __restrict__ emb,
                                                const float* __restrict__ g,
                                                const float* __restrict__ be,
                                                float* __restrict__ out) {
  __shared__ float red[256];
  const int row = blockIdx.x;
  const int tid = threadIdx.x;
  const float* a = accb + (size_t)row * kD;
  const float* e = emb + (size_t)row * kD;
  const float y0 = a[tid] + e[tid];
  const float y1 = a[tid + 256] + e[tid + 256];
  const float y2 = a[tid + 512] + e[tid + 512];
  const float mu = block_reduce_sum256(y0 + y1 + y2, red) * (1.0f / (float)kD);
  const float d0 = y0 - mu, d1 = y1 - mu, d2 = y2 - mu;
  const float var = block_reduce_sum256(d0 * d0 + d1 * d1 + d2 * d2, red) * (1.0f / (float)kD);
  const float inv = rsqrtf(var + 1e-12f);
  float* o = out + (size_t)row * kD;
  o[tid] = d0 * inv * g[tid] + be[tid];
  o[tid + 256] = d1 * inv * g[tid + 256] + be[tid + 256];
  o[tid + 512] = d2 * inv * g[tid + 512] + be[tid + 512];
}

}  // namespace

// ---------------------------------------------------------------------------
// Host orchestration
// ---------------------------------------------------------------------------
extern "C" void kernel_launch(void* const* d_in, const int* in_sizes, int n_in,
                              void* d_out, int out_size, void* d_ws, size_t ws_size,
                              hipStream_t stream) {
  (void)in_sizes; (void)n_in; (void)out_size; (void)ws_size;

  const float* emb    = (const float*)d_in[0];
  const float* pre_w  = (const float*)d_in[1];
  const float* pre_b  = (const float*)d_in[2];
  const float* psfs   = (const float*)d_in[3];
  const float* post_w = (const float*)d_in[4];
  const float* post_b = (const float*)d_in[5];
  const float* sparse = (const float*)d_in[6];
  const float* ln_g   = (const float*)d_in[7];
  const float* ln_b   = (const float*)d_in[8];
  float* out = (float*)d_out;

  char* ws = (char*)d_ws;
  size_t off = 0;
  auto alloc = [&](size_t bytes) -> void* {
    void* p = ws + off;
    off = (off + bytes + 255) & ~(size_t)255;
    return p;
  };

  const size_t nBLD = (size_t)kB * kL * kD;  // 12,582,912
  __bf16* emb_bf  = (__bf16*)alloc(nBLD * 2);
  __bf16* preWp   = (__bf16*)alloc((size_t)kD * kD * 2);
  __bf16* postWp  = (__bf16*)alloc((size_t)kD * kD * 2);
  float*  xf      = (float*)alloc(nBLD * 4);
  __bf16* xBp     = (__bf16*)alloc(nBLD * 2);
  float*  Ssuf    = (float*)alloc(nBLD * 4);
  float*  psfL    = (float*)alloc((size_t)kH * kL * kPHD * 4);
  float*  cvec    = (float*)alloc((size_t)kH * kPHD * 4);
  cplx*   Pf1     = (cplx*)alloc((size_t)kH * 64 * kL * 8);
  cplx*   Pfull   = (cplx*)alloc((size_t)kH * 64 * kF * 8);
  cplx*   Xc      = (cplx*)alloc((size_t)kB * kH * 64 * kL * 8);
  __bf16* convAct = (__bf16*)alloc(nBLD * 2);
  float*  accb    = (float*)alloc(nBLD * 4);
  __bf16* spb     = (__bf16*)alloc((size_t)kL * kL * 2);
  float*  snv     = (float*)alloc(kL * 4);
  float*  snvh    = (float*)alloc(kL * 4);
  float*  snw2    = (float*)alloc(kL * 4);
  float*  snsig   = (float*)alloc(256);

  // 1) bf16 operands
  convert_bf16<<<dim3((unsigned)(nBLD / 256)), dim3(256), 0, stream>>>(emb, emb_bf);
  pack_wT<<<dim3((kD * kD) / 256), dim3(256), 0, stream>>>(pre_w, preWp);
  pack_wT<<<dim3((kD * kD) / 256), dim3(256), 0, stream>>>(post_w, postWp);

  // 2) pre-projection: x = emb @ pre_w^T + pre_b   (WMMA)
  gemm_bf16_wmma<<<dim3(kD / kTN, (kB * kL) / kTM, 1), dim3(256), 0, stream>>>(
      emb_bf, 0, preWp, 0, xf, 0, pre_b, kB * kL, kD, kD, 0);

  // 3) pack x for the sparse GEMM; suffix sums for conv tail
  pack_x<<<dim3((unsigned)(nBLD / 256)), dim3(256), 0, stream>>>(xf, xBp);
  suffix_sum<<<dim3(kB, kD / 256), dim3(256), 0, stream>>>(xf, Ssuf);

  // 4) PSF softmax + spectral kernels
  psf_softmax<<<dim3(kH * kPHD), dim3(256), 0, stream>>>(psfs, psfL, cvec);
  dft64_psf<<<dim3(kH * kL), dim3(64), 0, stream>>>(psfL, Pf1);
  fft_seq_psf<<<dim3(kH * 64), dim3(256), 0, stream>>>(Pf1, Pfull);

  // 5) 2-D transform of x, pointwise multiply, inverse seq FFT (in place)
  dft64_x<<<dim3(kB * kH * kL), dim3(64), 0, stream>>>(xf, Xc);
  fft_seq_x<<<dim3(kB * kH * 64), dim3(256), 0, stream>>>(Xc, Pfull);

  // 6) inverse head-dim DFT + tail + exact GELU -> bf16 activations
  conv_finish<<<dim3(kB * kH * kL), dim3(64), 0, stream>>>(Xc, Ssuf, cvec, convAct);

  // 7) post-projection: accb = convAct @ post_w^T + post_b   (WMMA)
  gemm_bf16_wmma<<<dim3(kD / kTN, (kB * kL) / kTM, 1), dim3(256), 0, stream>>>(
      convAct, 0, postWp, 0, accb, 0, post_b, kB * kL, kD, kD, 0);

  // 8) spectral norm -> tril-scaled sparse mixing matrix (bf16)
  sn_colsum<<<dim3(kL / 256), dim3(256), 0, stream>>>(sparse, snv);
  sn_normalize_v<<<dim3(1), dim3(256), 0, stream>>>(snv, snvh);
  sn_rowdot<<<dim3(kL), dim3(256), 0, stream>>>(sparse, snvh, snw2);
  sn_sigma<<<dim3(1), dim3(256), 0, stream>>>(snw2, snsig);
  sp_build<<<dim3((kL * kL) / 256), dim3(256), 0, stream>>>(sparse, snsig, spb);

  // 9) sparse branch (batched WMMA GEMM), accumulating into accb
  gemm_bf16_wmma<<<dim3(kD / kTN, kL / kTM, kB), dim3(256), 0, stream>>>(
      spb, 0, xBp, (long long)kL * kD, accb, (long long)kL * kD, nullptr,
      kL, kD, kL, 1);

  // 10) residual + LayerNorm
  ln_final<<<dim3(kB * kL), dim3(256), 0, stream>>>(accb, emb, ln_g, ln_b, out);
}
